// DifferentiableEmbeddingClassifier_56693568307431
// MI455X (gfx1250) — matrix operand order repaired and checked by reading, not compiled, not run on hardware
//
#include <hip/hip_runtime.h>
#include <hip/hip_bf16.h>

// ---------------------------------------------------------------------------
// DifferentiableEmbeddingClassifier for MI455X (gfx1250, wave32, WMMA).
//
// out[t,v] = X_{sel(v)}[t,:] . (weight[:,v] * mask[v,:]) + bias[v]
//   X_b = x @ blk_w[b]^T + blk_b[b]
//   mask[v,d] = (d < gate_v) + frac(1e9*gate_v)*1e-9,  gate_v = gates_w[v]*1024
//   sel(v) = floor(gates_w[v] * 5 * (1 - 1e-10))
//
// phase1: X_b (b=0..4) via bf16 WMMA -> bf16 workspace (41.9 MB)
// phase2: stacked-K GEMM (K = 5*1024). Block = 64 tok x 128 vocab:
//         A chunks staged to LDS with GLOBAL_LOAD_ASYNC_TO_LDS_B128
//         (s_wait_asynccnt before the barrier); falls back to
//         global_load_b128 + ds_store_b128 if the builtin is absent.
//         Double-buffered LDS with padded stride (bank-conflict-free reads).
//         B fragment built in regs from weight*mask under sel(v)==b predicate,
//         4 accumulators/wave reuse each B fragment over 4 token sub-tiles.
// Memory-bound: 524 MB output write dominates; weight(131MB)+Xb(42MB) fit L2.
// ---------------------------------------------------------------------------

typedef __bf16 bf16_t;
typedef __attribute__((ext_vector_type(16))) __bf16 v16bf;
typedef __attribute__((ext_vector_type(8)))  float  v8f;
typedef __attribute__((ext_vector_type(4)))  int    v4i;

#if defined(__has_builtin)
#  if __has_builtin(__builtin_amdgcn_global_load_async_to_lds_b128)
#    define HAVE_ASYNC_LDS 1
#  endif
#  if __has_builtin(__builtin_amdgcn_s_wait_asynccnt)
#    define HAVE_WAIT_ASYNC 1
#  endif
#endif

#define TOKENS 4096   // 2 * 2048
#define DIMK   1024
#define VOCAB  32000
#define NBLK   5

#define TBLK   64     // tokens per block (phase 2)
#define LDA    40     // LDS row stride in halves (32 data + 8 pad -> 20 banks)

// -------------------------------- phase 1 ----------------------------------
// One wave computes one 16x16 tile of X_b = x @ blk_w[b]^T + blk_b[b].
__global__ __launch_bounds__(256)
void dec_phase1_block_gemm(const float* __restrict__ x,      // [TOKENS, DIMK]
                           const float* __restrict__ blk_w,  // [NBLK, DIMK, DIMK]
                           const float* __restrict__ blk_b,  // [NBLK, DIMK]
                           bf16_t* __restrict__ xb)          // [NBLK, TOKENS, DIMK]
{
    const int lane = threadIdx.x & 31;
    const int wave = threadIdx.x >> 5;
    const int tile = blockIdx.x * 8 + wave;

    const int dTiles = DIMK / 16;                  // 64
    const int tTiles = TOKENS / 16;                // 256
    const int b    = tile / (tTiles * dTiles);
    const int rem  = tile % (tTiles * dTiles);
    const int t0   = (rem / dTiles) * 16;
    const int d0   = (rem % dTiles) * 16;

    const int  col   = lane & 15;
    const bool hi    = (lane >= 16);
    const int  abase = hi ? 8  : 0;   // A-fragment K base (ISA 16-bit A layout)
    const int  bbase = hi ? 16 : 0;   // unused here: B is built with abase, like A

    const float* __restrict__ xrow = x + (size_t)(t0 + col) * DIMK;
    const float* __restrict__ wrow = blk_w + ((size_t)b * DIMK + (d0 + col)) * DIMK;

    v8f acc = {};
    for (int k0 = 0; k0 < DIMK; k0 += 32) {
        v16bf afrag, bfrag;
#pragma unroll
        for (int j = 0; j < 16; ++j) {
            const int kl = abase + j + ((j >= 8) ? 8 : 0);   // {base..base+7, base+16..}
            afrag[j] = (bf16_t)xrow[k0 + kl];
        }
#pragma unroll
        for (int j = 0; j < 16; ++j) {                       // same K order as A
            bfrag[j] = (bf16_t)wrow[k0 + abase + j + ((j >= 8) ? 8 : 0)];
        }
        acc = __builtin_amdgcn_wmma_f32_16x16x32_bf16(
                  false, afrag, false, bfrag, (short)0, acc, false, false);
    }

    const float bb = blk_b[b * DIMK + d0 + col];
#pragma unroll
    for (int i = 0; i < 8; ++i) {
        const int row = i + (hi ? 8 : 0);                    // C/D layout
        xb[((size_t)b * TOKENS + (t0 + row)) * DIMK + (d0 + col)] =
            (bf16_t)(acc[i] + bb);
    }
}

// -------------------------------- phase 2 ----------------------------------
// Block: 8 waves, covers TBLK(64) tokens x 128 vocab. Per K-chunk (32):
//   - all 256 threads stage the 64x32 bf16 A chunk to LDS (16 B each, async;
//     the copy overlaps the weight-fragment build)
//   - each wave builds its predicated B fragment from weight*mask
//   - each wave runs 4 WMMAs (4 token sub-tiles), A fragments from LDS
__global__ __launch_bounds__(256)
void dec_phase2_masked_gemm(const bf16_t* __restrict__ xb,      // [NBLK, TOKENS, DIMK]
                            const float*  __restrict__ gates_w, // [VOCAB]
                            const float*  __restrict__ weight,  // [DIMK, VOCAB]
                            const float*  __restrict__ bias,    // [VOCAB]
                            float*        __restrict__ out)     // [TOKENS, VOCAB]
{
    __shared__ __align__(16) bf16_t lsA[2][TBLK * LDA];   // 2 x 5 KB + pad

    const int tid  = threadIdx.x;
    const int lane = tid & 31;
    const int wave = tid >> 5;

    const int vGroups = VOCAB / 128;               // 250
    const int t0 = (blockIdx.x / vGroups) * TBLK;
    const int v0 = (blockIdx.x % vGroups) * 128 + wave * 16;

    const int  col   = lane & 15;
    const bool hi    = (lane >= 16);
    const int  abase = hi ? 8  : 0;
    const int  bbase = hi ? 16 : 0;

    const int   v    = v0 + col;
    const float gw   = gates_w[v];
    const float gate = gw * (float)DIMK;
    const int   sel  = (int)floorf(gw * (float)NBLK * (1.0f - 1e-10f));
    const float t1e9 = 1e9f * gate;
    const float fr   = (t1e9 - floorf(t1e9)) * 1e-9f;   // straight-through residual

    // cooperative staging: thread -> (row, 8-half column segment)
    const int sr = tid >> 2;          // 0..63
    const int sc = (tid & 3) * 8;     // 0,8,16,24

    v8f acc[4] = {v8f{}, v8f{}, v8f{}, v8f{}};

#pragma unroll 2
    for (int it = 0; it < (NBLK * DIMK) / 32; ++it) {
        const int k0    = it * 32;
        const int b     = k0 >> 10;              // chunk lies in one block b
        const int dbase = k0 & (DIMK - 1);
        const int p     = it & 1;

        // ---- stage A chunk: xb[b][t0..t0+63][dbase..dbase+31] -> LDS ----
        const uint4* gsrc = (const uint4*)
            &xb[((size_t)b * TOKENS + (t0 + sr)) * DIMK + (dbase + sc)];
        bf16_t* ldst = &lsA[p][sr * LDA + sc];
#if defined(HAVE_ASYNC_LDS)
        __builtin_amdgcn_global_load_async_to_lds_b128((v4i*)gsrc, (v4i*)ldst, 0, 0);
#else
        *(uint4*)ldst = *gsrc;
#endif

        // ---- build predicated B fragment in registers (overlaps the copy) --
        v16bf bfrag;
        if (sel == b) {
#pragma unroll
            for (int j = 0; j < 16; ++j) {
                const int   d = dbase + abase + j + ((j >= 8) ? 8 : 0);
                const float w = weight[(size_t)d * VOCAB + v];
                const float m = (((float)d < gate) ? 1.0f : 0.0f) + fr;
                bfrag[j] = (bf16_t)(w * m);
            }
            if (dbase + 64 <= DIMK)   // hint next chunk's weight lines
                __builtin_prefetch(&weight[(size_t)(dbase + bbase + 32) * VOCAB + v], 0, 1);
        } else {
#pragma unroll
            for (int j = 0; j < 16; ++j) bfrag[j] = (bf16_t)0.0f;
        }

#if defined(HAVE_ASYNC_LDS)
#  if defined(HAVE_WAIT_ASYNC)
        __builtin_amdgcn_s_wait_asynccnt(0);
#  else
        asm volatile("s_wait_asynccnt 0" ::: "memory");
#  endif
#endif
        __syncthreads();   // chunk visible to all waves (double buffer: 1 sync/iter)

        // ---- 4 token sub-tiles: A fragment from LDS (2 x ds_load_b128) ----
#pragma unroll
        for (int s = 0; s < 4; ++s) {
            const int row = s * 16 + col;
            const uint4* lp = (const uint4*)&lsA[p][row * LDA + abase];
            union { uint4 u[2]; v16bf v; } au;
            au.u[0] = lp[0];          // K = abase .. abase+7
            au.u[1] = lp[2];          // K = abase+16 .. abase+23
            acc[s] = __builtin_amdgcn_wmma_f32_16x16x32_bf16(
                         false, au.v, false, bfrag, (short)0, acc[s], false, false);
        }
    }

    const float bv = bias[v];
#pragma unroll
    for (int s = 0; s < 4; ++s) {
#pragma unroll
        for (int i = 0; i < 8; ++i) {
            const int row = s * 16 + i + (hi ? 8 : 0);
            out[(size_t)(t0 + row) * VOCAB + v] = acc[s][i] + bv;
        }
    }
}

// ------------------------------- launcher ----------------------------------
extern "C" void kernel_launch(void* const* d_in, const int* in_sizes, int n_in,
                              void* d_out, int out_size, void* d_ws, size_t ws_size,
                              hipStream_t stream) {
    const float* x       = (const float*)d_in[0];  // [2,2048,1024]
    const float* gates_w = (const float*)d_in[1];  // [32000,1]
    const float* weight  = (const float*)d_in[2];  // [1024,32000]
    const float* bias    = (const float*)d_in[3];  // [32000]
    const float* blk_w   = (const float*)d_in[4];  // [5,1024,1024]
    const float* blk_b   = (const float*)d_in[5];  // [5,1024]
    float* out = (float*)d_out;                    // [2,2048,32000]

    bf16_t* xb = (bf16_t*)d_ws;  // requires NBLK*TOKENS*DIMK*2 = 41,943,040 bytes

    const int tiles1 = NBLK * (TOKENS / 16) * (DIMK / 16);   // 81920
    dec_phase1_block_gemm<<<tiles1 / 8, 256, 0, stream>>>(x, blk_w, blk_b, xb);

    const int blocks2 = (TOKENS / TBLK) * (VOCAB / 128);     // 64 * 250 = 16000
    dec_phase2_masked_gemm<<<blocks2, 256, 0, stream>>>(xb, gates_w, weight, bias, out);
}
